// _SeqGenerator2D_54185307406556
// MI455X (gfx1250) — matrix operand order repaired and checked by reading, not compiled, not run on hardware
//
#include <hip/hip_runtime.h>
#include <hip/hip_bf16.h>

typedef _Float16 half_t;
typedef __attribute__((ext_vector_type(16))) _Float16 v16h;
typedef __attribute__((ext_vector_type(8)))  _Float16 v8h;
typedef __attribute__((ext_vector_type(8)))  float    v8f;

union V16 { v16h v; v8h h[2]; };

#define BB   8192
#define TT   50
#define SS   100
#define EE   64
#define HH   128
#define G4   512
#define KK   192
#define MM   256

// ---------------------------------------------------------------------------
// Prep: concat [Wih | Whh] -> f16 [512][192] for encoder and decoder
// ---------------------------------------------------------------------------
__global__ __launch_bounds__(256) void k_prep_w(
    const float* __restrict__ Wih_e, const float* __restrict__ Whh_e,
    const float* __restrict__ Wih_d, const float* __restrict__ Whh_d,
    half_t* __restrict__ Wcat_e, half_t* __restrict__ Wcat_d)
{
    int i = blockIdx.x * 256 + threadIdx.x;
    if (i >= G4 * KK) return;
    int n = i / KK, k = i % KK;
    float ve = (k < EE) ? Wih_e[n * EE + k] : Whh_e[n * HH + (k - EE)];
    float vd = (k < EE) ? Wih_d[n * EE + k] : Whh_d[n * HH + (k - EE)];
    Wcat_e[i] = (half_t)ve;
    Wcat_d[i] = (half_t)vd;
}

// ---------------------------------------------------------------------------
// Init: offsets = diff(obs, prepend 0), base = obs[:,T-1], h=c=0
// ---------------------------------------------------------------------------
__global__ __launch_bounds__(256) void k_init(
    const float* __restrict__ obs, float* __restrict__ offsets,
    float* __restrict__ base, float* __restrict__ h, float* __restrict__ c)
{
    long i = (long)blockIdx.x * 256 + threadIdx.x;
    if (i < (long)BB * TT * 2) {
        int comp = (int)(i & 1);
        long bt = i >> 1;
        int t = (int)(bt % TT);
        long b = bt / TT;
        float cur  = obs[i];
        float prev = (t > 0) ? obs[i - 2] : 0.f;
        offsets[i] = cur - prev;
        if (t == TT - 1) base[b * 2 + comp] = cur;   // telescoped cumsum of offsets
    }
    if (i < (long)BB * HH) { h[i] = 0.f; c[i] = 0.f; }
}

// ---------------------------------------------------------------------------
// Fused LSTM step: G = [x|h] @ Wcat^T (+bias), gates, in-place h/c update.
// Decoder mode additionally computes off = h2 @ Wr^T + br, cumsum, output.
// One block = 16 batch rows x all 512 gate columns. 8 waves x 4 N-tiles.
// ---------------------------------------------------------------------------
__global__ __launch_bounds__(256) void k_lstm_step(
    const float* __restrict__ offsets,     // [B][T][2]
    float* __restrict__ off_buf,           // [B][2] decoder autoregressive input
    const float* __restrict__ Wx,          // [64][2]  (We or Wd)
    const float* __restrict__ bx,          // [64]
    const half_t* __restrict__ Wcat,       // [512][192] f16 row-major
    const float* __restrict__ bias,        // [512]
    float* __restrict__ h, float* __restrict__ c,   // [B][128] in-place
    float* __restrict__ pos,               // [B][2] running cumsum
    float* __restrict__ out,               // [B][S][2]
    const float* __restrict__ Wr,          // [2][128]
    const float* __restrict__ br,          // [2]
    int t, int is_dec)
{
    __shared__ __attribute__((aligned(16))) half_t sA[16][208];  // [row][K=192]
    __shared__ float sG[16][520];                                // [row][512]
    __shared__ float sH[16][132];                                // decoder h2

    const int tid = threadIdx.x;
    const int r0  = blockIdx.x * 16;

    // ---- Phase 1: build A tile [16][192] = [embed(x) | h] in f16 ----
    for (int idx = tid; idx < 16 * EE; idx += 256) {
        int r = idx >> 6, k = idx & 63;
        int b = r0 + r;
        float ox, oy;
        if (is_dec) { ox = off_buf[b * 2 + 0]; oy = off_buf[b * 2 + 1]; }
        else        { ox = offsets[((long)b * TT + t) * 2 + 0];
                      oy = offsets[((long)b * TT + t) * 2 + 1]; }
        float x = Wx[k * 2 + 0] * ox + Wx[k * 2 + 1] * oy + bx[k];
        sA[r][k] = (half_t)x;
    }
    for (int idx = tid; idx < 16 * HH; idx += 256) {
        int r = idx >> 7, k = idx & 127;
        sA[r][EE + k] = (half_t)h[(long)(r0 + r) * HH + k];
    }
    __syncthreads();

    // ---- Phase 2: WMMA GEMM  G[16][512] = A[16][192] x Wcat^T ----
    {
        const int wave = tid >> 5, lane = tid & 31;
        const int laneLo = lane & 15, laneHi = lane >> 4;
        v8f acc[4] = {};
        #pragma unroll
        for (int kc = 0; kc < 6; ++kc) {
            V16 af;
            const half_t* ap = &sA[laneLo][kc * 32 + laneHi * 8];
            af.h[0] = *(const v8h*)(ap);
            af.h[1] = *(const v8h*)(ap + 16);
            #pragma unroll
            for (int nt = 0; nt < 4; ++nt) {
                int n = wave * 64 + nt * 16 + laneLo;
                const half_t* bp = Wcat + (long)n * KK + kc * 32 + laneHi * 8;
                V16 bf;
                bf.h[0] = *(const v8h*)(bp);
                bf.h[1] = *(const v8h*)(bp + 16);
                acc[nt] = __builtin_amdgcn_wmma_f32_16x16x32_f16(
                    false, af.v, false, bf.v, (short)0, acc[nt], false, false);
            }
        }
        // scatter C/D tiles to LDS: VGPR v -> row v + laneHi*8, col = laneLo
        #pragma unroll
        for (int nt = 0; nt < 4; ++nt) {
            int col = wave * 64 + nt * 16 + laneLo;
            #pragma unroll
            for (int v = 0; v < 8; ++v)
                sG[v + laneHi * 8][col] = acc[nt][v];
        }
    }
    __syncthreads();

    // ---- Phase 3: LSTM cell (gate order i,f,g,o) ----
    for (int idx = tid; idx < 16 * HH; idx += 256) {
        int r = idx >> 7, j = idx & 127;
        long b = r0 + r;
        float gi = sG[r][j]           + bias[j];
        float gf = sG[r][HH + j]      + bias[HH + j];
        float gg = sG[r][2 * HH + j]  + bias[2 * HH + j];
        float go = sG[r][3 * HH + j]  + bias[3 * HH + j];
        float cold = c[b * HH + j];
        float si = 1.f / (1.f + __expf(-gi));
        float sf = 1.f / (1.f + __expf(-gf));
        float so = 1.f / (1.f + __expf(-go));
        float c2 = sf * cold + si * tanhf(gg);
        float h2 = so * tanhf(c2);
        c[b * HH + j] = c2;
        h[b * HH + j] = h2;
        if (is_dec) sH[r][j] = h2;
    }

    // ---- Phase 4 (decoder): off = h2 @ Wr^T + br; cumsum; emit output ----
    if (is_dec) {
        __syncthreads();
        if (tid < 32) {
            int r = tid >> 1, comp = tid & 1;
            long b = r0 + r;
            float a = br[comp];
            #pragma unroll 8
            for (int k = 0; k < HH; ++k) a += sH[r][k] * Wr[comp * HH + k];
            float p = pos[b * 2 + comp] + a;
            pos[b * 2 + comp] = p;
            off_buf[b * 2 + comp] = a;
            out[(b * SS + t) * 2 + comp] = p;
        }
    }
}

// ---------------------------------------------------------------------------
// One-time MLP (tiny vs the recurrences): plain fp32
// ---------------------------------------------------------------------------
__global__ __launch_bounds__(256) void k_mlp1(
    const float* __restrict__ h, const float* __restrict__ Wm1,
    const float* __restrict__ bm1, float* __restrict__ out1)
{
    __shared__ float sh[HH];
    long b = blockIdx.x;
    int m = threadIdx.x;                      // 256 threads
    if (m < HH) sh[m] = h[b * HH + m];
    __syncthreads();
    float a = bm1[m];
    #pragma unroll 8
    for (int k = 0; k < HH; ++k) a += sh[k] * Wm1[m * HH + k];
    out1[b * MM + m] = fmaxf(a, 0.f);
}

__global__ __launch_bounds__(128) void k_mlp2_init_dec(
    const float* __restrict__ out1, const float* __restrict__ Wm2,
    const float* __restrict__ bm2,  const float* __restrict__ z,
    const float* __restrict__ base, const float* __restrict__ offsets,
    float* __restrict__ h, float* __restrict__ c,
    float* __restrict__ pos, float* __restrict__ off_buf)
{
    __shared__ float sm[MM];
    long b = blockIdx.x;
    int j = threadIdx.x;                      // 128 threads
    sm[j]       = out1[b * MM + j];
    sm[j + 128] = out1[b * MM + j + 128];
    __syncthreads();
    float v;
    if (j < 112) {
        float a = bm2[j];
        #pragma unroll 8
        for (int k = 0; k < MM; ++k) a += sm[k] * Wm2[j * MM + k];
        v = fmaxf(a, 0.f);
    } else {
        v = z[b * 16 + (j - 112)];
    }
    h[b * HH + j] = v;
    c[b * HH + j] = 0.f;
    if (j == 0) {
        pos[b * 2 + 0] = base[b * 2 + 0];
        pos[b * 2 + 1] = base[b * 2 + 1];
        off_buf[b * 2 + 0] = offsets[(b * TT + (TT - 1)) * 2 + 0];
        off_buf[b * 2 + 1] = offsets[(b * TT + (TT - 1)) * 2 + 1];
    }
}

// ---------------------------------------------------------------------------
extern "C" void kernel_launch(void* const* d_in, const int* in_sizes, int n_in,
                              void* d_out, int out_size, void* d_ws, size_t ws_size,
                              hipStream_t stream) {
    (void)in_sizes; (void)n_in; (void)out_size; (void)ws_size;

    const float* obs   = (const float*)d_in[0];
    // d_in[1] = num_steps (device scalar) -> compile-time SS=100
    const float* We    = (const float*)d_in[2];
    const float* be    = (const float*)d_in[3];
    const float* Wih_e = (const float*)d_in[4];
    const float* Whh_e = (const float*)d_in[5];
    const float* b_e   = (const float*)d_in[6];
    const float* Wm1   = (const float*)d_in[7];
    const float* bm1   = (const float*)d_in[8];
    const float* Wm2   = (const float*)d_in[9];
    const float* bm2   = (const float*)d_in[10];
    const float* Wd    = (const float*)d_in[11];
    const float* bd    = (const float*)d_in[12];
    const float* Wih_d = (const float*)d_in[13];
    const float* Whh_d = (const float*)d_in[14];
    const float* b_d   = (const float*)d_in[15];
    const float* Wr    = (const float*)d_in[16];
    const float* br    = (const float*)d_in[17];
    const float* z     = (const float*)d_in[18];
    float* out = (float*)d_out;

    // ---- workspace carve (256B aligned) ----
    char* p = (char*)d_ws;
    auto carve = [&](size_t bytes) -> char* {
        char* r = p;
        p += (bytes + 255) & ~(size_t)255;
        return r;
    };
    float*  offsets = (float*) carve((size_t)BB * TT * 2 * 4);
    float*  base    = (float*) carve((size_t)BB * 2 * 4);
    float*  h       = (float*) carve((size_t)BB * HH * 4);
    float*  c       = (float*) carve((size_t)BB * HH * 4);
    float*  out1    = (float*) carve((size_t)BB * MM * 4);
    float*  pos     = (float*) carve((size_t)BB * 2 * 4);
    float*  off_buf = (float*) carve((size_t)BB * 2 * 4);
    half_t* Wcat_e  = (half_t*)carve((size_t)G4 * KK * 2);
    half_t* Wcat_d  = (half_t*)carve((size_t)G4 * KK * 2);

    // ---- prep ----
    k_prep_w<<<(G4 * KK + 255) / 256, 256, 0, stream>>>(
        Wih_e, Whh_e, Wih_d, Whh_d, Wcat_e, Wcat_d);
    {
        long n1 = (long)BB * TT * 2, n2 = (long)BB * HH;
        long n = n1 > n2 ? n1 : n2;
        k_init<<<(int)((n + 255) / 256), 256, 0, stream>>>(obs, offsets, base, h, c);
    }

    // ---- encoder: 50 sequential fused LSTM steps ----
    for (int t = 0; t < TT; ++t)
        k_lstm_step<<<BB / 16, 256, 0, stream>>>(
            offsets, off_buf, We, be, Wcat_e, b_e, h, c,
            pos, out, Wr, br, t, 0);

    // ---- MLP + decoder state init ----
    k_mlp1<<<BB, 256, 0, stream>>>(h, Wm1, bm1, out1);
    k_mlp2_init_dec<<<BB, 128, 0, stream>>>(out1, Wm2, bm2, z, base, offsets,
                                            h, c, pos, off_buf);

    // ---- decoder: 100 autoregressive fused LSTM steps ----
    for (int s = 0; s < SS; ++s)
        k_lstm_step<<<BB / 16, 256, 0, stream>>>(
            offsets, off_buf, Wd, bd, Wcat_d, b_d, h, c,
            pos, out, Wr, br, s, 1);
}
